// RNNModel_55164559949961
// MI455X (gfx1250) — matrix operand order repaired and checked by reading, not compiled, not run on hardware
//
#include <hip/hip_runtime.h>

#define HID   1024
#define G3    3072
#define NB    64        // batch
#define SQ    512       // sequence
#define EMB   256
#define VOC   256

typedef __attribute__((ext_vector_type(16))) __bf16 v16bf;
typedef __attribute__((ext_vector_type(8)))  float  v8f;

union FragAB { v16bf v; unsigned int u[8]; };
union FragC  { v8f   v; float        f[8]; };

__device__ __forceinline__ unsigned short f2bf(float f) {
  unsigned int u = __float_as_uint(f);
  u += 0x7FFFu + ((u >> 16) & 1u);      // round-to-nearest-even
  return (unsigned short)(u >> 16);
}

__device__ __forceinline__ float sigmf(float x) { return 1.0f / (1.0f + __expf(-x)); }

// A fragment: 16x32 bf16, row m = lane%16, K layout per CDNA5 ISA 7.12.2:
// lanes 0-15: V0..V3 -> K=0..7, V4..V7 -> K=16..23 ; lanes 16-31: +8.
__device__ __forceinline__ v16bf load_a_frag(const unsigned short* __restrict__ A,
                                             int row, int stride, int koff, int hi) {
  FragAB f;
  const unsigned short* base = A + row * stride + koff + hi * 8;
  f.u[0] = *(const unsigned int*)(base + 0);
  f.u[1] = *(const unsigned int*)(base + 2);
  f.u[2] = *(const unsigned int*)(base + 4);
  f.u[3] = *(const unsigned int*)(base + 6);
  f.u[4] = *(const unsigned int*)(base + 16);
  f.u[5] = *(const unsigned int*)(base + 18);
  f.u[6] = *(const unsigned int*)(base + 20);
  f.u[7] = *(const unsigned int*)(base + 22);
  return f.v;
}

// B fragment: 32x16 bf16, col n = lane%16; lanes 0-15: V0..V3 -> K=0..7,
// V4..V7 -> K=16..23 ; lanes 16-31: +8 (same K layout as A fragment).
// WT stored as [N][K] (weight pre-transposed) so K is contiguous.
__device__ __forceinline__ v16bf load_b_frag(const unsigned short* __restrict__ WT,
                                             int col, int Kdim, int koff, int hi) {
  FragAB f;
  const unsigned short* base = WT + col * Kdim + koff + hi * 8;
#pragma unroll
  for (int v = 0; v < 8; ++v) f.u[v] = *(const unsigned int*)(base + (v < 4 ? 2 * v : 16 + 2 * (v - 4)));
  return f.v;
}

__device__ __forceinline__ v8f wmma_bf16(v16bf a, v16bf b, v8f c) {
  return __builtin_amdgcn_wmma_f32_16x16x32_bf16(false, a, false, b, (short)0, c,
                                                 false, false);
}

// ---------------- weight convert + transpose: dst[n*K+k] = bf16(src[k*N+n]) ---
__global__ void __launch_bounds__(256) transpose_bf16_kernel(
    const float* __restrict__ src, unsigned short* __restrict__ dst, int K, int N) {
  long long idx = (long long)blockIdx.x * 256 + threadIdx.x;
  if (idx >= (long long)K * N) return;
  int k = (int)(idx / N);
  int n = (int)(idx % N);
  dst[(long long)n * K + k] = f2bf(src[idx]);
}

// ---------------- embedding gather, fp32 -> bf16 -----------------------------
__global__ void __launch_bounds__(256) embed_kernel(
    const int* __restrict__ x, const float* __restrict__ emb,
    unsigned short* __restrict__ Eall) {
  int bs = blockIdx.x;            // 0..B*S-1
  int e  = threadIdx.x;           // 0..255
  int tok = x[bs];
  Eall[(long long)bs * EMB + e] = f2bf(emb[tok * EMB + e]);
}

__global__ void __launch_bounds__(256) zero_kernel(unsigned int* __restrict__ p, int n) {
  int i = blockIdx.x * 256 + threadIdx.x;
  if (i < n) p[i] = 0u;
}

// ---------------- one GRU timestep (both GEMMs + gates fused) ----------------
// gates[64,3072] = Ain[64,Kin] @ W + h_old[64,1024] @ U ; then Keras GRU update.
// 64 blocks x 128 threads = 256 waves; wave -> (m_tile in 0..3, ng in 0..63).
__global__ void __launch_bounds__(128) gru_step_kernel(
    const unsigned short* __restrict__ Ain, int strideA, int Kin,
    const unsigned short* __restrict__ WT,     // [3072][Kin]
    const unsigned short* __restrict__ UT,     // [3072][1024]
    const float* __restrict__ bx, const float* __restrict__ bh,
    const float* __restrict__ h_old_f32, const unsigned short* __restrict__ h_old_bf,
    float* __restrict__ h_new_f32, unsigned short* __restrict__ h_new_bf,
    unsigned short* __restrict__ out_all, long long strideOut) {
  const int tid = threadIdx.x;
  const int lane = tid & 31, wave = tid >> 5;
  const int lane_lo = lane & 15, hi = lane >> 4;
  const int gw = blockIdx.x * 4 + wave;     // 0..255
  const int m_tile = gw & 3;
  const int ng = gw >> 2;                   // 0..63
  const int arow = m_tile * 16 + lane_lo;   // A-matrix row (batch index for lane)
  const int nz = ng * 16, nr = HID + ng * 16, nh = 2 * HID + ng * 16;

  v8f acc_z = {}, acc_r = {}, acc_xh = {}, acc_hh = {};

  // input-side gates: A = Ain (Kin = 256 for layer0, 1024 for layer1)
  for (int ko = 0; ko < Kin; ko += 32) {
    v16bf a  = load_a_frag(Ain, arow, strideA, ko, hi);
    v16bf bz = load_b_frag(WT, nz + lane_lo, Kin, ko, hi);
    v16bf br = load_b_frag(WT, nr + lane_lo, Kin, ko, hi);
    v16bf bg = load_b_frag(WT, nh + lane_lo, Kin, ko, hi);
    acc_z  = wmma_bf16(a, bz, acc_z);
    acc_r  = wmma_bf16(a, br, acc_r);
    acc_xh = wmma_bf16(a, bg, acc_xh);
  }
  // recurrent gates: A = h_old (K = 1024)
  for (int ko = 0; ko < HID; ko += 32) {
    v16bf a  = load_a_frag(h_old_bf, arow, HID, ko, hi);
    v16bf bz = load_b_frag(UT, nz + lane_lo, HID, ko, hi);
    v16bf br = load_b_frag(UT, nr + lane_lo, HID, ko, hi);
    v16bf bg = load_b_frag(UT, nh + lane_lo, HID, ko, hi);
    acc_z  = wmma_bf16(a, bz, acc_z);
    acc_r  = wmma_bf16(a, br, acc_r);
    acc_hh = wmma_bf16(a, bg, acc_hh);
  }

  FragC cz, cr, cx, ch;
  cz.v = acc_z; cr.v = acc_r; cx.v = acc_xh; ch.v = acc_hh;

  const int col = nz + lane_lo;                 // hidden column in [0,1024)
  const float bzc = bx[nz + lane_lo] + bh[nz + lane_lo];
  const float brc = bx[nr + lane_lo] + bh[nr + lane_lo];
  const float bxh = bx[nh + lane_lo];
  const float bhh = bh[nh + lane_lo];

#pragma unroll
  for (int v = 0; v < 8; ++v) {
    const int m = m_tile * 16 + v + 8 * hi;     // batch row (C/D layout)
    float z  = sigmf(cz.f[v] + bzc);
    float r  = sigmf(cr.f[v] + brc);
    float hc = tanhf(cx.f[v] + bxh + r * (ch.f[v] + bhh));
    float ho = h_old_f32[m * HID + col];
    float hn = z * ho + (1.0f - z) * hc;
    h_new_f32[m * HID + col] = hn;
    unsigned short hb = f2bf(hn);
    h_new_bf[m * HID + col] = hb;
    if (out_all) out_all[(long long)m * strideOut + col] = hb;
  }
}

// ---------------- dense: logits[32768,256] = Hall @ Wd + bd ------------------
// 1024 blocks x 256 threads = 8192 waves; wave -> (m_tile 0..2047, ngrp 0..3).
__global__ void __launch_bounds__(256) dense_kernel(
    const unsigned short* __restrict__ Hall,   // [B*S][1024] bf16
    const unsigned short* __restrict__ WdT,    // [256][1024] bf16
    const float* __restrict__ bd, float* __restrict__ out) {
  const int tid = threadIdx.x;
  const int lane = tid & 31, wave = tid >> 5;
  const int lane_lo = lane & 15, hi = lane >> 4;
  const int gw = blockIdx.x * 8 + wave;      // 0..8191
  const int m_tile = gw >> 2;
  const int ngrp = gw & 3;                   // 4 n-tiles each -> 64 cols
  const int arow = m_tile * 16 + lane_lo;

  v8f acc0 = {}, acc1 = {}, acc2 = {}, acc3 = {};
  for (int ko = 0; ko < HID; ko += 32) {
    v16bf a  = load_a_frag(Hall, arow, HID, ko, hi);
    v16bf b0 = load_b_frag(WdT, ngrp * 64 +  0 + lane_lo, HID, ko, hi);
    v16bf b1 = load_b_frag(WdT, ngrp * 64 + 16 + lane_lo, HID, ko, hi);
    v16bf b2 = load_b_frag(WdT, ngrp * 64 + 32 + lane_lo, HID, ko, hi);
    v16bf b3 = load_b_frag(WdT, ngrp * 64 + 48 + lane_lo, HID, ko, hi);
    acc0 = wmma_bf16(a, b0, acc0);
    acc1 = wmma_bf16(a, b1, acc1);
    acc2 = wmma_bf16(a, b2, acc2);
    acc3 = wmma_bf16(a, b3, acc3);
  }
  FragC c0, c1, c2, c3;
  c0.v = acc0; c1.v = acc1; c2.v = acc2; c3.v = acc3;
#pragma unroll
  for (int v = 0; v < 8; ++v) {
    const int m = m_tile * 16 + v + 8 * hi;
    float* o = out + (long long)m * VOC + ngrp * 64 + lane_lo;
    o[ 0] = c0.f[v] + bd[ngrp * 64 +  0 + lane_lo];
    o[16] = c1.f[v] + bd[ngrp * 64 + 16 + lane_lo];
    o[32] = c2.f[v] + bd[ngrp * 64 + 32 + lane_lo];
    o[48] = c3.f[v] + bd[ngrp * 64 + 48 + lane_lo];
  }
}

// ---------------- softmax over 256 logits, one wave per row ------------------
__global__ void __launch_bounds__(256) softmax_kernel(float* __restrict__ out) {
  const int lane = threadIdx.x & 31, wave = threadIdx.x >> 5;
  const long long row = (long long)blockIdx.x * 8 + wave;   // 0..32767
  float* p = out + row * VOC;
  float vals[8];
  float mx = -3.4e38f;
#pragma unroll
  for (int i = 0; i < 8; ++i) { vals[i] = p[lane + i * 32]; mx = fmaxf(mx, vals[i]); }
#pragma unroll
  for (int off = 16; off > 0; off >>= 1) mx = fmaxf(mx, __shfl_xor(mx, off, 32));
  float s = 0.0f;
#pragma unroll
  for (int i = 0; i < 8; ++i) { vals[i] = __expf(vals[i] - mx); s += vals[i]; }
#pragma unroll
  for (int off = 16; off > 0; off >>= 1) s += __shfl_xor(s, off, 32);
  const float inv = 1.0f / s;
#pragma unroll
  for (int i = 0; i < 8; ++i) p[lane + i * 32] = vals[i] * inv;
}

// ---------------------------------------------------------------------------
extern "C" void kernel_launch(void* const* d_in, const int* in_sizes, int n_in,
                              void* d_out, int out_size, void* d_ws, size_t ws_size,
                              hipStream_t stream) {
  const int*   x    = (const int*)  d_in[0];
  const float* emb  = (const float*)d_in[1];
  const float* W0   = (const float*)d_in[2];
  const float* U0   = (const float*)d_in[3];
  const float* b0   = (const float*)d_in[4];
  const float* W1   = (const float*)d_in[5];
  const float* U1   = (const float*)d_in[6];
  const float* b1   = (const float*)d_in[7];
  const float* Wd   = (const float*)d_in[8];
  const float* bd   = (const float*)d_in[9];
  float*       out  = (float*)d_out;

  // ---- workspace carve (all 256B aligned) ----
  size_t off = 0;
  char* base = (char*)d_ws;
  auto carve = [&](size_t bytes) -> void* {
    void* p = base + off;
    off += (bytes + 255) & ~(size_t)255;
    return p;
  };
  unsigned short* W0T  = (unsigned short*)carve((size_t)G3 * EMB * 2);  // [3072][256]
  unsigned short* U0T  = (unsigned short*)carve((size_t)G3 * HID * 2);  // [3072][1024]
  unsigned short* W1T  = (unsigned short*)carve((size_t)G3 * HID * 2);
  unsigned short* U1T  = (unsigned short*)carve((size_t)G3 * HID * 2);
  unsigned short* WdT  = (unsigned short*)carve((size_t)VOC * HID * 2); // [256][1024]
  unsigned short* Eall = (unsigned short*)carve((size_t)NB * SQ * EMB * 2);
  unsigned short* Hall = (unsigned short*)carve((size_t)NB * SQ * HID * 2);
  // contiguous state region (zeroed below)
  char* states = (char*)carve(0);
  float*          h0f[2]; unsigned short* h0b[2];
  float*          h1f[2]; unsigned short* h1b[2];
  h0f[0] = (float*)carve((size_t)NB * HID * 4);
  h0f[1] = (float*)carve((size_t)NB * HID * 4);
  h1f[0] = (float*)carve((size_t)NB * HID * 4);
  h1f[1] = (float*)carve((size_t)NB * HID * 4);
  h0b[0] = (unsigned short*)carve((size_t)NB * HID * 2);
  h0b[1] = (unsigned short*)carve((size_t)NB * HID * 2);
  h1b[0] = (unsigned short*)carve((size_t)NB * HID * 2);
  h1b[1] = (unsigned short*)carve((size_t)NB * HID * 2);
  size_t state_bytes = (size_t)((char*)base + off - states);

  // ---- one-time conversions (per call; deterministic) ----
  {
    long long n;
    n = (long long)EMB * G3;
    transpose_bf16_kernel<<<(unsigned)((n + 255) / 256), 256, 0, stream>>>(W0, W0T, EMB, G3);
    n = (long long)HID * G3;
    transpose_bf16_kernel<<<(unsigned)((n + 255) / 256), 256, 0, stream>>>(U0, U0T, HID, G3);
    transpose_bf16_kernel<<<(unsigned)((n + 255) / 256), 256, 0, stream>>>(W1, W1T, HID, G3);
    transpose_bf16_kernel<<<(unsigned)((n + 255) / 256), 256, 0, stream>>>(U1, U1T, HID, G3);
    n = (long long)HID * VOC;
    transpose_bf16_kernel<<<(unsigned)((n + 255) / 256), 256, 0, stream>>>(Wd, WdT, HID, VOC);
  }
  embed_kernel<<<NB * SQ, 256, 0, stream>>>(x, emb, Eall);
  {
    int nwords = (int)(state_bytes / 4);
    zero_kernel<<<(nwords + 255) / 256, 256, 0, stream>>>((unsigned int*)states, nwords);
  }

  // ---- sequential recurrence: 512 steps x 2 layers ----
  const float* b0x = b0;           // input bias row
  const float* b0h = b0 + G3;      // recurrent bias row
  const float* b1x = b1;
  const float* b1h = b1 + G3;
  int p = 0;
  for (int t = 0; t < SQ; ++t) {
    // layer 0: A = embedded input at step t (row stride = SQ*EMB per batch)
    gru_step_kernel<<<64, 128, 0, stream>>>(
        Eall + (long long)t * EMB, SQ * EMB, EMB,
        W0T, U0T, b0x, b0h,
        h0f[p], h0b[p], h0f[1 - p], h0b[1 - p],
        (unsigned short*)nullptr, 0LL);
    // layer 1: A = h0 just produced; also record h1 into Hall for the dense GEMM
    gru_step_kernel<<<64, 128, 0, stream>>>(
        h0b[1 - p], HID, HID,
        W1T, U1T, b1x, b1h,
        h1f[p], h1b[p], h1f[1 - p], h1b[1 - p],
        Hall + (long long)t * HID, (long long)SQ * HID);
    p ^= 1;
  }

  // ---- dense logits + softmax (TEMPERATURE == 1.0) ----
  dense_kernel<<<1024, 256, 0, stream>>>(Hall, WdT, bd, out);
  softmax_kernel<<<NB * SQ / 8, 256, 0, stream>>>(out);
}
